// CustomLoss_38852274159738
// MI455X (gfx1250) — compile-verified, matrix operand order checked by reading, not run on hardware
//
#include <hip/hip_runtime.h>

namespace {
constexpr int K = 512;
constexpr int O = 8;
constexpr int NS = 3;
constexpr int PLANE = K * K;        // 262144
constexpr int KKO = K * K * O;      // 2097152
constexpr int KC = 64;              // k-chunk per LDS tile
constexpr int LP = 72;              // LDS pitch in halves (pad vs 64)
}

typedef _Float16 half_t;
typedef __attribute__((ext_vector_type(8)))  _Float16 v8h;
typedef __attribute__((ext_vector_type(16))) _Float16 v16h;
typedef __attribute__((ext_vector_type(4)))  float v4f;
typedef __attribute__((ext_vector_type(8)))  float v8f;

// Fragment load: ISA 16-bit 16x32 layout => per lane two contiguous 8-half
// chunks at k-offsets (8*halfSel) and (16 + 8*halfSel) within the 32-wide step.
__device__ __forceinline__ v16h ld_frag(const half_t* rowBase, int kb) {
  v8h c0 = *(const v8h*)(rowBase + kb);
  v8h c1 = *(const v8h*)(rowBase + kb + 16);
  return __builtin_shufflevector(c0, c1, 0, 1, 2, 3, 4, 5, 6, 7,
                                 8, 9, 10, 11, 12, 13, 14, 15);
}

#define WMMA_F16(a, b, c)                                                     \
  __builtin_amdgcn_wmma_f32_16x16x32_f16(false, (a), false, (b), (short)0,    \
                                         (c), false, false)

// ---------------------------------------------------------------------------
// Stage 0: fold P-scalings, split fp32 -> (hi,lo) f16 pair, emit GEMM layouts:
//   Ahi/Alo[x][j][k]  (k contiguous)   from A[j,k,x]*Pb[j]*Pc[k]
//   BhiT/BloT[y][i][k] (k contiguous)  from B[k,i,y]*Pa[i]   (pre-transposed)
// ---------------------------------------------------------------------------
__global__ __launch_bounds__(256) void kl_prep(const float* __restrict__ yp,
                                               half_t* __restrict__ Ahi,
                                               half_t* __restrict__ Alo,
                                               half_t* __restrict__ BhiT,
                                               half_t* __restrict__ BloT) {
  const float* Pa = yp;
  const float* Pb = yp + K;
  const float* Pc = yp + 2 * K;
  const float* A = yp + NS * K;
  const float* B = A + KKO;

  const int bid = blockIdx.x;
  const bool doB = bid >= 1024;
  const int lin = ((doB ? bid - 1024 : bid) << 8) + threadIdx.x;  // 0..262143
  const int r = lin >> 9;
  const int c = lin & (K - 1);

  if (!doB) {  // lin = j*512 + k
    const float s = Pb[r] * Pc[c];
    const float* src = A + (size_t)lin * O;  // 8 contiguous x values
#pragma unroll
    for (int x = 0; x < O; ++x) {
      const float v = src[x] * s;
      const half_t h = (half_t)v;
      Ahi[(size_t)x * PLANE + lin] = h;
      Alo[(size_t)x * PLANE + lin] = (half_t)(v - (float)h);
    }
  } else {     // lin = i*512 + k  (transposed output)
    const int i = r, k = c;
    const float s = Pa[i];
    const float* src = B + ((size_t)k * K + i) * O;  // 8 contiguous y values
#pragma unroll
    for (int y = 0; y < O; ++y) {
      const float v = src[y] * s;
      const half_t h = (half_t)v;
      BhiT[(size_t)y * PLANE + lin] = h;
      BloT[(size_t)y * PLANE + lin] = (half_t)(v - (float)h);
    }
  }
}

// ---------------------------------------------------------------------------
// Stage 1: 64 GEMMs M_xy = Aw_x * Bw_y via split-f16 WMMA (hh + hl + lh),
// fp32 accumulate, fused C-contraction epilogue.
// Grid: 2048 blocks = xy(64) x jt(8) x it(4); 256 threads = 8 waves;
// wave tile 32x32 (2x2 fp32 16x16 accumulators); double-buffered LDS.
// ---------------------------------------------------------------------------
__global__ __launch_bounds__(256) void kl_gemm(const half_t* __restrict__ Ahi,
                                               const half_t* __restrict__ Alo,
                                               const half_t* __restrict__ BhiT,
                                               const half_t* __restrict__ BloT,
                                               const float* __restrict__ C,
                                               float* __restrict__ partials) {
  __shared__ half_t AsH[2][64][LP];
  __shared__ half_t AsL[2][64][LP];
  __shared__ half_t BsH[2][128][LP];
  __shared__ half_t BsL[2][128][LP];
  __shared__ float zred[8][8];

  const int bid = blockIdx.x;
  const int xy = bid >> 5;
  const int jt = (bid >> 2) & 7;
  const int itile = bid & 3;
  const int x = xy >> 3, y = xy & 7;
  const int jBase = jt * 64;
  const int iBase = itile * 128;

  const int tid = threadIdx.x;
  const int lane = tid & 31;
  const int wave = tid >> 5;
  const int wj = wave >> 2;            // 0..1
  const int wi = wave & 3;             // 0..3
  const int halfSel = (lane >> 4) & 1;
  const int l16 = lane & 15;

  const half_t* AH = Ahi + (size_t)x * PLANE;
  const half_t* AL = Alo + (size_t)x * PLANE;
  const half_t* BH = BhiT + (size_t)y * PLANE;
  const half_t* BL = BloT + (size_t)y * PLANE;

  // cooperative staging coordinates
  const int aRow = tid >> 2;           // 0..63
  const int aCol = (tid & 3) * 16;     // halves: 0,16,32,48
  const int bRow = tid >> 1;           // 0..127
  const int bCol = (tid & 1) * 32;     // halves: 0,32

  v8f acc00 = {}, acc01 = {}, acc10 = {}, acc11 = {};

  {  // stage tile 0
    const size_t aOfs = (size_t)(jBase + aRow) * K + aCol;
    const size_t bOfs = (size_t)(iBase + bRow) * K + bCol;
    *(v8h*)&AsH[0][aRow][aCol]      = *(const v8h*)(AH + aOfs);
    *(v8h*)&AsH[0][aRow][aCol + 8]  = *(const v8h*)(AH + aOfs + 8);
    *(v8h*)&AsL[0][aRow][aCol]      = *(const v8h*)(AL + aOfs);
    *(v8h*)&AsL[0][aRow][aCol + 8]  = *(const v8h*)(AL + aOfs + 8);
    *(v8h*)&BsH[0][bRow][bCol]      = *(const v8h*)(BH + bOfs);
    *(v8h*)&BsH[0][bRow][bCol + 8]  = *(const v8h*)(BH + bOfs + 8);
    *(v8h*)&BsH[0][bRow][bCol + 16] = *(const v8h*)(BH + bOfs + 16);
    *(v8h*)&BsH[0][bRow][bCol + 24] = *(const v8h*)(BH + bOfs + 24);
    *(v8h*)&BsL[0][bRow][bCol]      = *(const v8h*)(BL + bOfs);
    *(v8h*)&BsL[0][bRow][bCol + 8]  = *(const v8h*)(BL + bOfs + 8);
    *(v8h*)&BsL[0][bRow][bCol + 16] = *(const v8h*)(BL + bOfs + 16);
    *(v8h*)&BsL[0][bRow][bCol + 24] = *(const v8h*)(BL + bOfs + 24);
  }
  __syncthreads();

  for (int itk = 0; itk < K / KC; ++itk) {
    const int cur = itk & 1;
    const bool hasNext = (itk + 1) < (K / KC);
    const int kn = (itk + 1) * KC;

    // prefetch next tile into registers (overlaps with WMMA below)
    v8h rAH0, rAH1, rAL0, rAL1;
    v8h rBH0, rBH1, rBH2, rBH3, rBL0, rBL1, rBL2, rBL3;
    if (hasNext) {
      const size_t aOfs = (size_t)(jBase + aRow) * K + kn + aCol;
      const size_t bOfs = (size_t)(iBase + bRow) * K + kn + bCol;
      rAH0 = *(const v8h*)(AH + aOfs);      rAH1 = *(const v8h*)(AH + aOfs + 8);
      rAL0 = *(const v8h*)(AL + aOfs);      rAL1 = *(const v8h*)(AL + aOfs + 8);
      rBH0 = *(const v8h*)(BH + bOfs);      rBH1 = *(const v8h*)(BH + bOfs + 8);
      rBH2 = *(const v8h*)(BH + bOfs + 16); rBH3 = *(const v8h*)(BH + bOfs + 24);
      rBL0 = *(const v8h*)(BL + bOfs);      rBL1 = *(const v8h*)(BL + bOfs + 8);
      rBL2 = *(const v8h*)(BL + bOfs + 16); rBL3 = *(const v8h*)(BL + bOfs + 24);
    }

    // compute on current buffer: 2 k-steps of 32, 12 WMMA each
#pragma unroll
    for (int ks = 0; ks < KC; ks += 32) {
      const int kb = ks + 8 * halfSel;
      v16h aH0 = ld_frag(&AsH[cur][wj * 32 + l16][0], kb);
      v16h aH1 = ld_frag(&AsH[cur][wj * 32 + 16 + l16][0], kb);
      v16h aL0 = ld_frag(&AsL[cur][wj * 32 + l16][0], kb);
      v16h aL1 = ld_frag(&AsL[cur][wj * 32 + 16 + l16][0], kb);
      v16h bH0 = ld_frag(&BsH[cur][wi * 32 + l16][0], kb);
      v16h bH1 = ld_frag(&BsH[cur][wi * 32 + 16 + l16][0], kb);
      v16h bL0 = ld_frag(&BsL[cur][wi * 32 + l16][0], kb);
      v16h bL1 = ld_frag(&BsL[cur][wi * 32 + 16 + l16][0], kb);
      // distance-4 between dependent ops on the same accumulator
      acc00 = WMMA_F16(aH0, bH0, acc00);
      acc01 = WMMA_F16(aH0, bH1, acc01);
      acc10 = WMMA_F16(aH1, bH0, acc10);
      acc11 = WMMA_F16(aH1, bH1, acc11);
      acc00 = WMMA_F16(aH0, bL0, acc00);
      acc01 = WMMA_F16(aH0, bL1, acc01);
      acc10 = WMMA_F16(aH1, bL0, acc10);
      acc11 = WMMA_F16(aH1, bL1, acc11);
      acc00 = WMMA_F16(aL0, bH0, acc00);
      acc01 = WMMA_F16(aL0, bH1, acc01);
      acc10 = WMMA_F16(aL1, bH0, acc10);
      acc11 = WMMA_F16(aL1, bH1, acc11);
    }

    if (hasNext) {
      const int nxt = cur ^ 1;
      *(v8h*)&AsH[nxt][aRow][aCol]      = rAH0;
      *(v8h*)&AsH[nxt][aRow][aCol + 8]  = rAH1;
      *(v8h*)&AsL[nxt][aRow][aCol]      = rAL0;
      *(v8h*)&AsL[nxt][aRow][aCol + 8]  = rAL1;
      *(v8h*)&BsH[nxt][bRow][bCol]      = rBH0;
      *(v8h*)&BsH[nxt][bRow][bCol + 8]  = rBH1;
      *(v8h*)&BsH[nxt][bRow][bCol + 16] = rBH2;
      *(v8h*)&BsH[nxt][bRow][bCol + 24] = rBH3;
      *(v8h*)&BsL[nxt][bRow][bCol]      = rBL0;
      *(v8h*)&BsL[nxt][bRow][bCol + 8]  = rBL1;
      *(v8h*)&BsL[nxt][bRow][bCol + 16] = rBL2;
      *(v8h*)&BsL[nxt][bRow][bCol + 24] = rBL3;
    }
    __syncthreads();
  }

  // ---- fused epilogue: partial_z = sum_{tile} M[j,i] * C[i,j,z] ----
  float zacc[8];
#pragma unroll
  for (int z = 0; z < 8; ++z) zacc[z] = 0.f;

  const int jW = jBase + wj * 32;
  const int iW = iBase + wi * 32;

#pragma unroll
  for (int ra = 0; ra < 2; ++ra) {
#pragma unroll
    for (int rb = 0; rb < 2; ++rb) {
      const v8f* accp = ra == 0 ? (rb == 0 ? &acc00 : &acc01)
                                : (rb == 0 ? &acc10 : &acc11);
      const int iG = iW + rb * 16 + l16;                  // N = lane%16
#pragma unroll
      for (int r = 0; r < 8; ++r) {
        const int jG = jW + ra * 16 + halfSel * 8 + r;    // M = r (+8 hi half)
        const float m = (*accp)[r];
        const float* crow = C + ((size_t)iG * K + jG) * O;
        v4f c0 = *(const v4f*)crow;
        v4f c1 = *(const v4f*)(crow + 4);
        zacc[0] += m * c0.x; zacc[1] += m * c0.y;
        zacc[2] += m * c0.z; zacc[3] += m * c0.w;
        zacc[4] += m * c1.x; zacc[5] += m * c1.y;
        zacc[6] += m * c1.z; zacc[7] += m * c1.w;
      }
    }
  }

#pragma unroll
  for (int z = 0; z < 8; ++z) {
    float v = zacc[z];
    for (int off = 16; off; off >>= 1) v += __shfl_xor(v, off, 32);
    if (lane == 0) zred[wave][z] = v;
  }
  __syncthreads();
  if (tid < 8) {
    float s = 0.f;
#pragma unroll
    for (int w = 0; w < 8; ++w) s += zred[w][tid];
    partials[(size_t)bid * 8 + tid] = s;   // deterministic: no atomics
  }
}

// ---------------------------------------------------------------------------
// Stage 2: reduce tile partials, emit probs + KL scalar.
// ---------------------------------------------------------------------------
__global__ __launch_bounds__(512) void kl_finalize(const float* __restrict__ partials,
                                                   const float* __restrict__ y_true,
                                                   float* __restrict__ out) {
  __shared__ float red[16];
  const int t = threadIdx.x;            // 0..511 == x*64 + y*8 + z
  const int xy = t >> 3, z = t & 7;

  float p = 0.f;
  for (int w = 0; w < 32; ++w) p += partials[(size_t)((xy << 5) + w) * 8 + z];
  out[1 + t] = p;

  const float yt = y_true[t];
  const float lp = logf(fminf(fmaxf(p, 1e-10f), 1.0f));
  float term = yt * (logf(yt + 1e-10f) - lp);

  for (int off = 16; off; off >>= 1) term += __shfl_xor(term, off, 32);
  if ((t & 31) == 0) red[t >> 5] = term;
  __syncthreads();
  if (t == 0) {
    float d = 0.f;
#pragma unroll
    for (int w = 0; w < 16; ++w) d += red[w];
    out[0] = d;
  }
}

extern "C" void kernel_launch(void* const* d_in, const int* in_sizes, int n_in,
                              void* d_out, int out_size, void* d_ws, size_t ws_size,
                              hipStream_t stream) {
  (void)in_sizes; (void)n_in; (void)out_size; (void)ws_size;
  const float* y_pred = (const float*)d_in[0];
  const float* y_true = (const float*)d_in[1];
  float* out = (float*)d_out;

  // ws layout: 4 f16 planes of KKO halves (4 MB each), then fp32 partials
  half_t* Ahi = (half_t*)d_ws;
  half_t* Alo = Ahi + (size_t)KKO;
  half_t* BhiT = Alo + (size_t)KKO;
  half_t* BloT = BhiT + (size_t)KKO;
  float* partials = (float*)(BloT + (size_t)KKO);

  const float* C = y_pred + NS * K + 2 * (size_t)KKO;

  kl_prep<<<2048, 256, 0, stream>>>(y_pred, Ahi, Alo, BhiT, BloT);
  kl_gemm<<<2048, 256, 0, stream>>>(Ahi, Alo, BhiT, BloT, C, partials);
  kl_finalize<<<1, 512, 0, stream>>>(partials, y_true, out);
}
